// DABlock_ours_90237262889331
// MI455X (gfx1250) — compile-verified, matrix operand order checked by reading, not run on hardware
//
#include <hip/hip_runtime.h>
#include <hip/hip_bf16.h>
#include <stdint.h>

typedef __attribute__((ext_vector_type(16))) __bf16 v16bf;
typedef __attribute__((ext_vector_type(8)))  __bf16 v8bf;
typedef __attribute__((ext_vector_type(8)))  float  v8f;
typedef int v4i_vs __attribute__((vector_size(16)));

#define BATCH 2
#define HW    4096
#define CIN   2048
#define CI    512
#define C8    64

// -------- gfx1250 async global->LDS copy (guarded; fallback = load + ds_store) ----
#if defined(__has_builtin)
#  if __has_builtin(__builtin_amdgcn_global_load_async_to_lds_b128)
#    define ASYNC_LDS 1
#  endif
#  if __has_builtin(__builtin_amdgcn_s_wait_asynccnt)
#    define HAVE_WAIT_ASYNC_BUILTIN 1
#  endif
#endif
#ifndef ASYNC_LDS
#  define ASYNC_LDS 0
#endif

__device__ __forceinline__ void stage16(const __bf16* src, __bf16* dst_lds) {
#if ASYNC_LDS
  __builtin_amdgcn_global_load_async_to_lds_b128(
      (__attribute__((address_space(1))) v4i_vs*)(uintptr_t)src,
      (__attribute__((address_space(3))) v4i_vs*)(uintptr_t)dst_lds, 0, 0);
#else
  *(v8bf*)dst_lds = *(const v8bf*)src;
#endif
}

__device__ __forceinline__ void wait_async() {
#if ASYNC_LDS
#  if defined(HAVE_WAIT_ASYNC_BUILTIN)
  __builtin_amdgcn_s_wait_asynccnt(0);
#  else
  asm volatile("s_wait_asynccnt 0x0" ::: "memory");
#  endif
#endif
}

// ---------- fragment helpers (CDNA5 16-bit WMMA VGPR layout) ----------
// A (16x32 bf16): lane L covers row M=L%16. lanes 0-15: K = {0..7,16..23},
// lanes 16-31: K = {8..15,24..31}. Element i<8 -> K=base+i ; i>=8 -> base+16+(i-8).
__device__ __forceinline__ int kidx(int lane, int i) {
  return ((i < 8) ? i : (8 + i)) + ((lane & 16) ? 8 : 0);
}

__device__ __forceinline__ v16bf bf_combine(const __bf16* p) {
  v8bf a = *(const v8bf*)p;
  v8bf b = *(const v8bf*)(p + 16);
  v16bf r;
#pragma unroll
  for (int i = 0; i < 8; ++i) { r[i] = a[i]; r[i + 8] = b[i]; }
  return r;
}

__device__ __forceinline__ v16bf bf_zero16() {
  v16bf r;
#pragma unroll
  for (int i = 0; i < 16; ++i) r[i] = (__bf16)0.0f;
  return r;
}

// ---------- small prep kernels ----------
__global__ void k_bn_prep(const float* s, const float* b, const float* m,
                          const float* v, float* inv, float* beta, int C) {
  int i = blockIdx.x * blockDim.x + threadIdx.x;
  if (i >= C) return;
  float iv = s[i] * rsqrtf(v[i] + 1e-5f);
  inv[i] = iv;
  beta[i] = b[i] - m[i] * iv;
}

// x: f32 NCHW [2,2048,64,64] -> xb: bf16 NHWC [B*HW, CIN]
__global__ void k_cvt_x(const float* __restrict__ x, __bf16* __restrict__ xb) {
  size_t i = (size_t)blockIdx.x * blockDim.x + threadIdx.x;
  int c = (int)(i & (CIN - 1));
  size_t row = i >> 11;
  int b = (int)(row >> 12);
  int p = (int)(row & (HW - 1));
  xb[i] = (__bf16)x[((size_t)(b * CIN + c) << 12) + p];
}

// Pack weights into exact per-lane WMMA B-fragment order:
// o = ((kt*tiles_n + nt)*32 + lane)*16 + i  <->  (k = kt*32+kidx(lane,i), n = nt*16+lane%16)
__global__ void k_pack_w(const float* __restrict__ w, __bf16* __restrict__ wp,
                         int Cic, int Co, int ksz) {
  size_t o = (size_t)blockIdx.x * blockDim.x + threadIdx.x;
  int K = ksz * ksz * Cic;
  if (o >= (size_t)K * Co) return;
  int i    = (int)(o & 15);
  int lane = (int)((o >> 4) & 31);
  size_t rest = o >> 9;
  int tiles_n = Co >> 4;
  int nt = (int)(rest % tiles_n);
  int kt = (int)(rest / tiles_n);
  int k = kt * 32 + kidx(lane, i);
  int n = nt * 16 + (lane & 15);
  int pos = k / Cic;
  int ci  = k - pos * Cic;
  float val;
  if (ksz == 3) val = w[(size_t)n * Cic * 9 + (size_t)ci * 9 + pos];
  else          val = w[(size_t)n * Cic + ci];
  wp[o] = (__bf16)val;
}

__global__ void k_transpose(const __bf16* __restrict__ in, __bf16* __restrict__ out,
                            int rows, int cols) {
  size_t i = (size_t)blockIdx.x * blockDim.x + threadIdx.x;
  if (i >= (size_t)rows * cols) return;
  int n = (int)(i / cols);
  int c = (int)(i - (size_t)n * cols);
  out[(size_t)c * rows + n] = in[i];
}

// ---------- implicit-GEMM conv with WMMA bf16, block-cooperative LDS B-staging ----
#define EPI_BF16_NHWC      0
#define EPI_F32_NCHW_STORE 1
#define EPI_F32_NCHW_ADD   2

template <int Cic, int Co, int KSZ>
__global__ __launch_bounds__(256)
void k_conv_wmma(const __bf16* __restrict__ act, const __bf16* __restrict__ wp,
                 const float* __restrict__ scale, const float* __restrict__ bias,
                 int relu, int mode,
                 __bf16* __restrict__ out_bf, float* __restrict__ out_f) {
  constexpr int tiles_n64 = (Co >> 6) ? (Co >> 6) : 1; // Co=64 -> 1
  constexpr int tiles_n   = Co >> 4;
  constexpr int K   = KSZ * KSZ * Cic;
  constexpr int nk  = K >> 5;
  constexpr int spt = Cic >> 5;      // K-steps per kernel tap (pow2)

  __shared__ __align__(32) __bf16 lb[2][4 * 32 * 16]; // 2 x 4KB

  const int tid  = threadIdx.x;
  const int lane = tid & 31;
  const int wave = tid >> 5;
  const int nt64 = blockIdx.x % tiles_n64;
  const int mt8  = blockIdx.x / tiles_n64;
  const int mt   = mt8 * 8 + wave;

  const int hi = lane >> 4;
  const int ml = lane & 15;
  const int r  = mt * 16 + ml;       // global row: b*HW + p
  const int bt = r >> 12;
  const int p  = r & (HW - 1);
  const int y  = p >> 6, xx = p & 63;

  const __bf16* wbase = wp + (size_t)(nt64 << 2) * 32 * 16;
  auto stage = [&](int kt, int s) {
    const __bf16* src = wbase + (size_t)kt * tiles_n * 512 + tid * 8;
    stage16(src, &lb[s][tid * 8]);
  };

  v8f acc[4] = {};
  const v16bf zf = bf_zero16();

  stage(0, 0);
  for (int kt = 0; kt < nk; ++kt) {
    const int cur = kt & 1;
    const int pos = kt / spt;              // shift (spt pow2)
    const int ci0 = (kt % spt) << 5;
    v16bf afrag;
    if constexpr (KSZ == 3) {
      int iy = y  + pos / 3 - 1;
      int ix = xx + pos % 3 - 1;
      if ((unsigned)iy < 64u && (unsigned)ix < 64u)
        afrag = bf_combine(act + ((size_t)((bt << 12) + (iy << 6) + ix)) * Cic
                               + ci0 + (hi << 3));
      else
        afrag = zf;
    } else {
      afrag = bf_combine(act + (size_t)r * Cic + ci0 + (hi << 3));
    }
    wait_async();
    __syncthreads();
    if (kt + 1 < nk) stage(kt + 1, cur ^ 1);
#pragma unroll
    for (int j = 0; j < 4; ++j) {
      v16bf bfrag = *(const v16bf*)&lb[cur][(j * 32 + lane) * 16];
      acc[j] = __builtin_amdgcn_wmma_f32_16x16x32_bf16(
          false, afrag, false, bfrag, (short)0, acc[j], false, false);
    }
  }

#pragma unroll
  for (int r8 = 0; r8 < 8; ++r8) {
    int row = mt * 16 + r8 + (hi << 3);
    int bb  = row >> 12;
    int pp  = row & (HW - 1);
#pragma unroll
    for (int j = 0; j < 4; ++j) {
      int ch = (nt64 << 6) + (j << 4) + ml;
      float val = acc[j][r8];
      if (scale) val *= scale[ch];
      if (bias)  val += bias[ch];
      if (relu)  val = fmaxf(val, 0.0f);
      if (mode == EPI_BF16_NHWC) {
        out_bf[(size_t)row * Co + ch] = (__bf16)val;
      } else {
        size_t oi = ((size_t)(bb * Co + ch) << 12) + pp; // NCHW
        if (mode == EPI_F32_NCHW_STORE) out_f[oi] = val;
        else                            out_f[oi] += val;
      }
    }
  }
}

// ---------- GEMM out[M,N] = A[M,K]*B[N,K]^T, LDS-staged B, double-buffered ----------
template <int N, int K>
__global__ __launch_bounds__(256)
void k_gemm_abt(const __bf16* __restrict__ A, const __bf16* __restrict__ Bm,
                float* __restrict__ out) {
  constexpr int tiles_n64 = N >> 6;
  constexpr int nk = K >> 5;
  __shared__ __align__(32) __bf16 lb[2][64 * 32]; // 64 rows x 32 K-elems = 4KB

  const int tid  = threadIdx.x;
  const int lane = tid & 31;
  const int wave = tid >> 5;
  const int nt64 = blockIdx.x % tiles_n64;
  const int mt8  = blockIdx.x / tiles_n64;
  const int mt   = mt8 * 8 + wave;

  const int hi = lane >> 4;
  const int ml = lane & 15;
  const __bf16* arow  = A  + (size_t)(mt * 16 + ml) * K + (hi << 3);
  const __bf16* bbase = Bm + (size_t)(nt64 * 64) * K;

  auto stage = [&](int kt, int s) {
    int row = tid >> 2, ch = tid & 3;
    const __bf16* src = bbase + (size_t)row * K + (kt << 5) + ch * 8;
    stage16(src, &lb[s][row * 32 + ch * 8]);
  };

  v8f acc[4] = {};
  stage(0, 0);
  for (int kt = 0; kt < nk; ++kt) {
    const int cur = kt & 1;
    const __bf16* ap = arow + ((size_t)kt << 5);
    __builtin_prefetch(ap + 64, 0, 1);
    v16bf afrag = bf_combine(ap);
    wait_async();
    __syncthreads();
    if (kt + 1 < nk) stage(kt + 1, cur ^ 1);
#pragma unroll
    for (int j = 0; j < 4; ++j) {
      v16bf bfrag = bf_combine(&lb[cur][(j * 16 + ml) * 32 + (hi << 3)]);
      acc[j] = __builtin_amdgcn_wmma_f32_16x16x32_bf16(
          false, afrag, false, bfrag, (short)0, acc[j], false, false);
    }
  }

#pragma unroll
  for (int r8 = 0; r8 < 8; ++r8) {
    int row = mt * 16 + r8 + (hi << 3);
#pragma unroll
    for (int j = 0; j < 4; ++j) {
      int col = nt64 * 64 + j * 16 + ml;
      out[(size_t)row * N + col] = acc[j][r8];
    }
  }
}

// ---------- softmax kernels ----------
__global__ void k_softmax_max(const float* __restrict__ E, __bf16* __restrict__ A, int N) {
  __shared__ float sd[256];
  const float* e = E + (size_t)blockIdx.x * N;
  __bf16* a = A + (size_t)blockIdx.x * N;
  float mx = -3.0e38f;
  for (int i = threadIdx.x; i < N; i += 256) mx = fmaxf(mx, e[i]);
  sd[threadIdx.x] = mx; __syncthreads();
  for (int s = 128; s > 0; s >>= 1) {
    if (threadIdx.x < s) sd[threadIdx.x] = fmaxf(sd[threadIdx.x], sd[threadIdx.x + s]);
    __syncthreads();
  }
  mx = sd[0]; __syncthreads();
  float sum = 0.0f;
  for (int i = threadIdx.x; i < N; i += 256) sum += __expf(e[i] - mx);
  sd[threadIdx.x] = sum; __syncthreads();
  for (int s = 128; s > 0; s >>= 1) {
    if (threadIdx.x < s) sd[threadIdx.x] += sd[threadIdx.x + s];
    __syncthreads();
  }
  float inv = 1.0f / sd[0];
  for (int i = threadIdx.x; i < N; i += 256) a[i] = (__bf16)(__expf(e[i] - mx) * inv);
}

// CAM: softmax of (rowmax - e) over d  ==  exp(rowmin - e)/sum (stable)
__global__ void k_softmax_min(const float* __restrict__ E, __bf16* __restrict__ A, int N) {
  __shared__ float sd[256];
  const float* e = E + (size_t)blockIdx.x * N;
  __bf16* a = A + (size_t)blockIdx.x * N;
  float mn = 3.0e38f;
  for (int i = threadIdx.x; i < N; i += 256) mn = fminf(mn, e[i]);
  sd[threadIdx.x] = mn; __syncthreads();
  for (int s = 128; s > 0; s >>= 1) {
    if (threadIdx.x < s) sd[threadIdx.x] = fminf(sd[threadIdx.x], sd[threadIdx.x + s]);
    __syncthreads();
  }
  mn = sd[0]; __syncthreads();
  float sum = 0.0f;
  for (int i = threadIdx.x; i < N; i += 256) sum += __expf(mn - e[i]);
  sd[threadIdx.x] = sum; __syncthreads();
  for (int s = 128; s > 0; s >>= 1) {
    if (threadIdx.x < s) sd[threadIdx.x] += sd[threadIdx.x + s];
    __syncthreads();
  }
  float inv = 1.0f / sd[0];
  for (int i = threadIdx.x; i < N; i += 256) a[i] = (__bf16)(__expf(mn - e[i]) * inv);
}

// dst[n,c] = bf16(gamma * src + feat[n,c]); src = gout[n*C+c] (trans=0) or gout[c*HW+n]
__global__ void k_resid(const float* __restrict__ gout, const __bf16* __restrict__ feat,
                        const float* __restrict__ gamma, __bf16* __restrict__ dst,
                        int C, int trans) {
  size_t i = (size_t)blockIdx.x * blockDim.x + threadIdx.x;
  if (i >= (size_t)HW * C) return;
  int n = (int)(i / C);
  int c = (int)(i - (size_t)n * C);
  float g = gamma[0];
  float s = trans ? gout[(size_t)c * HW + n] : gout[i];
  dst[i] = (__bf16)(g * s + (float)feat[i]);
}

// ---------- host orchestration ----------
extern "C" void kernel_launch(void* const* d_in, const int* in_sizes, int n_in,
                              void* d_out, int out_size, void* d_ws, size_t ws_size,
                              hipStream_t stream) {
  (void)in_sizes; (void)n_in; (void)out_size; (void)ws_size;
  const float* x     = (const float*)d_in[0];
  const float* w5a   = (const float*)d_in[1];
  const float* bn5as = (const float*)d_in[2];
  const float* bn5ab = (const float*)d_in[3];
  const float* bn5am = (const float*)d_in[4];
  const float* bn5av = (const float*)d_in[5];
  const float* w5c   = (const float*)d_in[6];
  const float* bn5cs = (const float*)d_in[7];
  const float* bn5cb = (const float*)d_in[8];
  const float* bn5cm = (const float*)d_in[9];
  const float* bn5cv = (const float*)d_in[10];
  const float* wq    = (const float*)d_in[11];
  const float* bq    = (const float*)d_in[12];
  const float* wk    = (const float*)d_in[13];
  const float* bk    = (const float*)d_in[14];
  const float* wv    = (const float*)d_in[15];
  const float* bv    = (const float*)d_in[16];
  const float* gpam  = (const float*)d_in[17];
  const float* gcam  = (const float*)d_in[18];
  const float* w51   = (const float*)d_in[19];
  const float* bn51s = (const float*)d_in[20];
  const float* bn51b = (const float*)d_in[21];
  const float* bn51m = (const float*)d_in[22];
  const float* bn51v = (const float*)d_in[23];
  const float* w52   = (const float*)d_in[24];
  const float* bn52s = (const float*)d_in[25];
  const float* bn52b = (const float*)d_in[26];
  const float* bn52m = (const float*)d_in[27];
  const float* bn52v = (const float*)d_in[28];
  float* out = (float*)d_out;

  char* base = (char*)d_ws;
  size_t off = 0;
  auto alloc = [&](size_t bytes) -> void* {
    void* p = base + off;
    off = (off + bytes + 255) & ~(size_t)255;
    return p;
  };

  __bf16* xb   = (__bf16*)alloc((size_t)BATCH * HW * CIN * 2);
  __bf16* w5aP = (__bf16*)alloc((size_t)9 * CIN * CI * 2);
  __bf16* w5cP = (__bf16*)alloc((size_t)9 * CIN * CI * 2);
  __bf16* w51P = (__bf16*)alloc((size_t)9 * CI * CI * 2);
  __bf16* w52P = (__bf16*)alloc((size_t)9 * CI * CI * 2);
  __bf16* wqP  = (__bf16*)alloc((size_t)CI * C8 * 2);
  __bf16* wkP  = (__bf16*)alloc((size_t)CI * C8 * 2);
  __bf16* wvP  = (__bf16*)alloc((size_t)CI * CI * 2);
  float* inv5a = (float*)alloc(CI * 4); float* beta5a = (float*)alloc(CI * 4);
  float* inv5c = (float*)alloc(CI * 4); float* beta5c = (float*)alloc(CI * 4);
  float* inv51 = (float*)alloc(CI * 4); float* beta51 = (float*)alloc(CI * 4);
  float* inv52 = (float*)alloc(CI * 4); float* beta52 = (float*)alloc(CI * 4);
  __bf16* feat1 = (__bf16*)alloc((size_t)BATCH * HW * CI * 2);
  __bf16* feat2 = (__bf16*)alloc((size_t)BATCH * HW * CI * 2);
  __bf16* qb    = (__bf16*)alloc((size_t)BATCH * HW * C8 * 2);
  __bf16* kb    = (__bf16*)alloc((size_t)BATCH * HW * C8 * 2);
  __bf16* vb    = (__bf16*)alloc((size_t)BATCH * HW * CI * 2);
  __bf16* vT    = (__bf16*)alloc((size_t)CI * HW * 2);
  __bf16* f2T   = (__bf16*)alloc((size_t)CI * HW * 2);
  float*  energy= (float*)alloc((size_t)HW * HW * 4);
  __bf16* attn  = (__bf16*)alloc((size_t)HW * HW * 2);
  float*  gout  = (float*)alloc((size_t)HW * CI * 4);
  __bf16* saf   = (__bf16*)alloc((size_t)BATCH * HW * CI * 2);
  __bf16* scf   = (__bf16*)alloc((size_t)BATCH * HW * CI * 2);

  // 1) conversion + packing + BN folding
  k_cvt_x<<<(BATCH * CIN * HW) / 256, 256, 0, stream>>>(x, xb);
  k_pack_w<<<(9 * CIN * CI) / 256, 256, 0, stream>>>(w5a, w5aP, CIN, CI, 3);
  k_pack_w<<<(9 * CIN * CI) / 256, 256, 0, stream>>>(w5c, w5cP, CIN, CI, 3);
  k_pack_w<<<(9 * CI * CI) / 256, 256, 0, stream>>>(w51, w51P, CI, CI, 3);
  k_pack_w<<<(9 * CI * CI) / 256, 256, 0, stream>>>(w52, w52P, CI, CI, 3);
  k_pack_w<<<(CI * C8) / 256, 256, 0, stream>>>(wq, wqP, CI, C8, 1);
  k_pack_w<<<(CI * C8) / 256, 256, 0, stream>>>(wk, wkP, CI, C8, 1);
  k_pack_w<<<(CI * CI) / 256, 256, 0, stream>>>(wv, wvP, CI, CI, 1);
  k_bn_prep<<<2, 256, 0, stream>>>(bn5as, bn5ab, bn5am, bn5av, inv5a, beta5a, CI);
  k_bn_prep<<<2, 256, 0, stream>>>(bn5cs, bn5cb, bn5cm, bn5cv, inv5c, beta5c, CI);
  k_bn_prep<<<2, 256, 0, stream>>>(bn51s, bn51b, bn51m, bn51v, inv51, beta51, CI);
  k_bn_prep<<<2, 256, 0, stream>>>(bn52s, bn52b, bn52m, bn52v, inv52, beta52, CI);

  // grids: (tiles_m/8) * tiles_n64 ; tiles_m = BATCH*HW/16 = 512 -> mt8 = 64
  const int g_c512 = 64 * (CI / 64);  // 512 blocks
  const int g_c64  = 64 * 1;          // 64 blocks

  // 2) heavy convs
  k_conv_wmma<CIN, CI, 3><<<g_c512, 256, 0, stream>>>(xb, w5aP, inv5a, beta5a, 1,
                                                      EPI_BF16_NHWC, feat1, nullptr);
  k_conv_wmma<CIN, CI, 3><<<g_c512, 256, 0, stream>>>(xb, w5cP, inv5c, beta5c, 1,
                                                      EPI_BF16_NHWC, feat2, nullptr);

  // 3) 1x1 projections
  k_conv_wmma<CI, C8, 1><<<g_c64, 256, 0, stream>>>(feat1, wqP, nullptr, bq, 0,
                                                    EPI_BF16_NHWC, qb, nullptr);
  k_conv_wmma<CI, C8, 1><<<g_c64, 256, 0, stream>>>(feat1, wkP, nullptr, bk, 0,
                                                    EPI_BF16_NHWC, kb, nullptr);
  k_conv_wmma<CI, CI, 1><<<g_c512, 256, 0, stream>>>(feat1, wvP, nullptr, bv, 0,
                                                     EPI_BF16_NHWC, vb, nullptr);

  // 4) attention, per batch (reuse energy/attn/gout scratch)
  for (int bb = 0; bb < BATCH; ++bb) {
    const __bf16* qBp = qb + (size_t)bb * HW * C8;
    const __bf16* kBp = kb + (size_t)bb * HW * C8;
    const __bf16* vBp = vb + (size_t)bb * HW * CI;
    const __bf16* f1p = feat1 + (size_t)bb * HW * CI;
    const __bf16* f2p = feat2 + (size_t)bb * HW * CI;

    // PAM: energy = q k^T ; attn = softmax ; out = attn vT^T
    k_gemm_abt<HW, C8><<<(HW / 128) * (HW / 64), 256, 0, stream>>>(qBp, kBp, energy);
    k_softmax_max<<<HW, 256, 0, stream>>>(energy, attn, HW);
    k_transpose<<<(HW * CI) / 256, 256, 0, stream>>>(vBp, vT, HW, CI);
    k_gemm_abt<CI, HW><<<(HW / 128) * (CI / 64), 256, 0, stream>>>(attn, vT, gout);
    k_resid<<<(HW * CI) / 256, 256, 0, stream>>>(gout, f1p, gpam,
                                                 saf + (size_t)bb * HW * CI, CI, 0);
    // CAM: energy = F F^T ; attn = softmax(rowmax-e) ; out = attn F
    k_transpose<<<(HW * CI) / 256, 256, 0, stream>>>(f2p, f2T, HW, CI);
    k_gemm_abt<CI, HW><<<(CI / 128) * (CI / 64), 256, 0, stream>>>(f2T, f2T, energy);
    k_softmax_min<<<CI, 256, 0, stream>>>(energy, attn, CI);
    k_gemm_abt<HW, CI><<<(CI / 128) * (HW / 64), 256, 0, stream>>>(attn, f2p, gout);
    k_resid<<<(HW * CI) / 256, 256, 0, stream>>>(gout, f2p, gcam,
                                                 scf + (size_t)bb * HW * CI, CI, 1);
  }

  // 5) final convs: out = cbr(saf,w51) + cbr(scf,w52), fp32 NCHW
  k_conv_wmma<CI, CI, 3><<<g_c512, 256, 0, stream>>>(saf, w51P, inv51, beta51, 1,
                                                     EPI_F32_NCHW_STORE, nullptr, out);
  k_conv_wmma<CI, CI, 3><<<g_c512, 256, 0, stream>>>(scf, w52P, inv52, beta52, 1,
                                                     EPI_F32_NCHW_ADD, nullptr, out);
}
